// FullAttention_35631048687907
// MI455X (gfx1250) — matrix operand order repaired and checked by reading, not compiled, not run on hardware
//
#include <hip/hip_runtime.h>

#define Bn 4
#define Tn 4096
#define En 2048
#define Dn 128

typedef __attribute__((ext_vector_type(16))) __bf16 v16bf;
typedef __attribute__((ext_vector_type(8)))  __bf16 v8bf;
typedef __attribute__((ext_vector_type(4)))  __bf16 v4bf;
typedef __attribute__((ext_vector_type(8)))  float  v8f;
typedef __attribute__((ext_vector_type(4)))  int    v4i;

#if defined(__AMDGCN__) && __has_builtin(__builtin_amdgcn_global_load_async_to_lds_b128)
#define HAVE_ASYNC_LDS 1
#else
#define HAVE_ASYNC_LDS 0
#endif

typedef __attribute__((address_space(1))) v4i g_v4i;  // global int4*
typedef __attribute__((address_space(3))) v4i l_v4i;  // LDS int4*

// 16-byte global -> LDS copy; async (ASYNCcnt-tracked) when available.
__device__ __forceinline__ void cp16_g2l(const __bf16* g, __bf16* l) {
#if HAVE_ASYNC_LDS
  __builtin_amdgcn_global_load_async_to_lds_b128(
      (g_v4i*)(g), (l_v4i*)(l), 0, 0);
#else
  *(v8bf*)l = *(const v8bf*)g;
#endif
}

__device__ __forceinline__ void wait_async_all() {
#if HAVE_ASYNC_LDS
#if __has_builtin(__builtin_amdgcn_s_wait_asynccnt)
  __builtin_amdgcn_s_wait_asynccnt(0);
#else
  asm volatile("s_wait_asynccnt 0x0" ::: "memory");
#endif
#endif
}

__device__ __forceinline__ v16bf cat8(v8bf a, v8bf b) {
  return __builtin_shufflevector(a, b, 0,1,2,3,4,5,6,7,8,9,10,11,12,13,14,15);
}

// A-fragment (16x32 bf16) from row-major row pointer (already at this lane's M row, K base).
// Half h covers the contiguous K run {h*16 .. h*16+15} (matches B-fragment order).
__device__ __forceinline__ v16bf load_a_frag(const __bf16* row, int h) {
  v8bf lo = *(const v8bf*)(row + h * 16);
  v8bf hi = *(const v8bf*)(row + h * 16 + 8);
  return cat8(lo, hi);
}

// B-fragment (32x16 bf16) from row-major row pointer (already at this lane's N row, K base).
// Same order as A-fragment: half h covers the contiguous K run {h*16 .. h*16+15}.
__device__ __forceinline__ v16bf load_b_frag(const __bf16* row, int h) {
  return *(const v16bf*)(row + h * 16);
}

__device__ __forceinline__ float redmax16(float v) {
  v = fmaxf(v, __shfl_xor(v, 1));
  v = fmaxf(v, __shfl_xor(v, 2));
  v = fmaxf(v, __shfl_xor(v, 4));
  v = fmaxf(v, __shfl_xor(v, 8));
  return v;
}
__device__ __forceinline__ float redsum16(float v) {
  v += __shfl_xor(v, 1);
  v += __shfl_xor(v, 2);
  v += __shfl_xor(v, 4);
  v += __shfl_xor(v, 8);
  return v;
}

// ---------------------------------------------------------------------------
// Kernel 1: fp32 -> bf16 conversion (vectorized x4)
// ---------------------------------------------------------------------------
__global__ __launch_bounds__(256) void cvt_f32_to_bf16(
    const float* __restrict__ in, unsigned short* __restrict__ out_, int n4) {
  v4bf* out = (v4bf*)out_;
  int i = blockIdx.x * blockDim.x + threadIdx.x;
  if (i < n4) {
    const float4 f = ((const float4*)in)[i];
    v4bf o = { (__bf16)f.x, (__bf16)f.y, (__bf16)f.z, (__bf16)f.w };
    out[i] = o;
  }
}

// ---------------------------------------------------------------------------
// Kernel 2: QKV projection. One wave computes a 16x128 tile of q, k, or v.
// The shared W K-panel (128x32 bf16, 8KB) is staged in LDS via async
// global->LDS copies, double buffered; copy of panel k+1 overlaps WMMAs of
// panel k. All 8 B-fragments are batch-loaded to registers, then the 8 WMMAs
// issue back-to-back. q,k row-major bf16 [B*T, D]; v transposed vT[b][d][t].
// ---------------------------------------------------------------------------
#define WPAD 40  // LDS row stride (bf16 elems): 32 data + 8 pad, keeps 16B align
__global__ __launch_bounds__(256) void qkv_gemm_kernel(
    const unsigned short* __restrict__ xb_,
    const unsigned short* __restrict__ wq_,
    const unsigned short* __restrict__ wk_,
    const unsigned short* __restrict__ wv_,
    unsigned short* __restrict__ qb_,
    unsigned short* __restrict__ kb_,
    unsigned short* __restrict__ vtb_) {
  __shared__ __bf16 Wl[2][128 * WPAD];

  const __bf16* xb = (const __bf16*)xb_;
  const int tid  = threadIdx.x;
  const int wave = tid >> 5;
  const int lane = tid & 31;
  const int lr   = lane & 15;
  const int h    = lane >> 4;
  const int tile  = blockIdx.x * 8 + wave;   // [0, Bn*Tn/16)
  const int which = blockIdx.y;              // 0=q, 1=k, 2=v (uniform per block)
  const __bf16* W = (const __bf16*)(which == 0 ? wq_ : which == 1 ? wk_ : wv_);

  // This thread's two 16B slices of the 8KB W panel (512 segs, 256 threads).
  const int rowA = tid >> 2;                 // rows 0..63
  const int colA = (tid & 3) * 8;            // col offset in elems (16B segs)
  const int rowB = rowA + 64;                // rows 64..127

  const int row0 = tile * 16;
  const __bf16* arow = xb + (size_t)(row0 + lr) * En;

  // Prologue: issue async copy of panel kk=0 into buffer 0.
  cp16_g2l(W + (size_t)rowA * En + colA, &Wl[0][rowA * WPAD + colA]);
  cp16_g2l(W + (size_t)rowB * En + colA, &Wl[0][rowB * WPAD + colA]);

  v8f acc[8] = {};
  for (int kk = 0; kk < En; kk += 32) {
    const int cur = (kk >> 5) & 1;
    wait_async_all();                        // own slice of panel 'cur' landed
    __syncthreads();                         // all waves' slices landed
    if (kk + 32 < En) {                      // prefetch next panel (other buf)
      const int nxt = cur ^ 1;
      const __bf16* g = W + kk + 32;
      cp16_g2l(g + (size_t)rowA * En + colA, &Wl[nxt][rowA * WPAD + colA]);
      cp16_g2l(g + (size_t)rowB * En + colA, &Wl[nxt][rowB * WPAD + colA]);
    }
    v16bf af = load_a_frag(arow + kk, h);
    const __bf16* wl = &Wl[cur][0];
    v16bf bfr[8];
#pragma unroll
    for (int n = 0; n < 8; ++n)              // batch all LDS fragment loads
      bfr[n] = *(const v16bf*)(wl + (n * 16 + lr) * WPAD + h * 16);
#pragma unroll
    for (int n = 0; n < 8; ++n)              // back-to-back WMMA chain
      acc[n] = __builtin_amdgcn_wmma_f32_16x16x32_bf16(
          false, af, false, bfr[n], (short)0, acc[n], false, false);
  }

  if (which == 2) {
    __bf16* vtb = (__bf16*)vtb_;
    const int b  = tile >> 8;                // Tn/16 = 256 tiles per batch
    const int t0 = (tile & 255) * 16;
#pragma unroll
    for (int n = 0; n < 8; ++n)
#pragma unroll
      for (int r = 0; r < 8; ++r)
        vtb[((size_t)b * Dn + n * 16 + lr) * Tn + t0 + r + 8 * h] = (__bf16)acc[n][r];
  } else {
    __bf16* ob = (__bf16*)(which == 0 ? qb_ : kb_);
#pragma unroll
    for (int n = 0; n < 8; ++n)
#pragma unroll
      for (int r = 0; r < 8; ++r)
        ob[(size_t)(row0 + r + 8 * h) * Dn + n * 16 + lr] = (__bf16)acc[n][r];
  }
}

// ---------------------------------------------------------------------------
// Kernel 3: flash attention, one wave32 per 16-query tile, 32-key blocks.
// ---------------------------------------------------------------------------
__global__ __launch_bounds__(32) void flash_attn_kernel(
    const unsigned short* __restrict__ qb_,
    const unsigned short* __restrict__ kb_,
    const unsigned short* __restrict__ vtb_,
    float* __restrict__ out) {
  const __bf16* qb  = (const __bf16*)qb_;
  const __bf16* kb  = (const __bf16*)kb_;
  const __bf16* vtb = (const __bf16*)vtb_;

  constexpr int PS = 40;                 // padded row stride (bf16 elems) for P tile
  __shared__ __bf16 Pl[16 * PS];

  const int tile = blockIdx.x;
  const int b  = tile >> 8;
  const int q0 = (tile & 255) * 16;
  const int lane = threadIdx.x & 31;
  const int lr = lane & 15;
  const int h  = lane >> 4;

  // Preload the 4 Q A-fragments covering D=128 (K-steps of 32).
  const __bf16* qrow = qb + ((size_t)b * Tn + q0 + lr) * Dn;
  v16bf qa[4];
#pragma unroll
  for (int s = 0; s < 4; ++s) qa[s] = load_a_frag(qrow + s * 32, h);

  v8f o[8] = {};
  float m[8], l[8];
#pragma unroll
  for (int r = 0; r < 8; ++r) { m[r] = -1e30f; l[r] = 0.0f; }

  const float sc = 1.0f / 128.0f;        // reference: softmax(scores / D)
  const int kend = q0 + 16;

  for (int k0 = 0; k0 < kend; k0 += 32) {
    const __bf16* kr0 = kb + ((size_t)b * Tn + k0 + lr) * Dn;
    const __bf16* kr1 = kr0 + 16 * Dn;
    if (k0 + 32 < kend) __builtin_prefetch(kr0 + 32 * Dn, 0, 0);

    // Batch-load all K B-fragments, then 8 back-to-back WMMAs:
    // S = Q @ K^T for two 16x16 column tiles (j in [k0,k0+16) / [k0+16,k0+32))
    v16bf kf0[4], kf1[4];
#pragma unroll
    for (int s = 0; s < 4; ++s) {
      kf0[s] = load_b_frag(kr0 + s * 32, h);
      kf1[s] = load_b_frag(kr1 + s * 32, h);
    }
    v8f s0 = {}, s1 = {};
#pragma unroll
    for (int s = 0; s < 4; ++s) {
      s0 = __builtin_amdgcn_wmma_f32_16x16x32_bf16(false, qa[s], false, kf0[s], (short)0, s0, false, false);
      s1 = __builtin_amdgcn_wmma_f32_16x16x32_bf16(false, qa[s], false, kf1[s], (short)0, s1, false, false);
    }

    // Issue V B-fragment loads now; their latency hides under the softmax VALU.
    v16bf vf[8];
#pragma unroll
    for (int n = 0; n < 8; ++n)
      vf[n] = load_b_frag(vtb + ((size_t)b * Dn + n * 16 + lr) * Tn + k0, h);

    // Online softmax. C-layout: lane holds col j=lr of rows (r + 8*h).
    float alpha[8];
#pragma unroll
    for (int r = 0; r < 8; ++r) {
      const int ig = q0 + r + 8 * h;     // global (per-batch) query row
      float a = s0[r] * sc;
      float c = s1[r] * sc;
      if (k0 + lr > ig)      a = -1e30f; // causal mask
      if (k0 + 16 + lr > ig) c = -1e30f;
      float mx = redmax16(fmaxf(a, c));
      mx = fmaxf(mx, m[r]);
      alpha[r] = __expf(m[r] - mx);
      m[r] = mx;
      float p0 = __expf(a - mx);
      float p1 = __expf(c - mx);
      l[r] = l[r] * alpha[r] + redsum16(p0 + p1);
      const int row = r + 8 * h;
      Pl[row * PS + lr]      = (__bf16)p0;
      Pl[row * PS + 16 + lr] = (__bf16)p1;
    }
    __syncthreads();                     // single-wave WG: orders LDS store->load

    // P tile back as A-fragment (16x32); rescale O, then 8 back-to-back WMMAs.
    v16bf pf = load_a_frag(&Pl[lr * PS], h);
#pragma unroll
    for (int n = 0; n < 8; ++n)
#pragma unroll
      for (int r = 0; r < 8; ++r) o[n][r] *= alpha[r];
#pragma unroll
    for (int n = 0; n < 8; ++n)
      o[n] = __builtin_amdgcn_wmma_f32_16x16x32_bf16(false, pf, false, vf[n], (short)0, o[n], false, false);
    __syncthreads();
  }

  float inv[8];
#pragma unroll
  for (int r = 0; r < 8; ++r) inv[r] = 1.0f / l[r];
#pragma unroll
  for (int n = 0; n < 8; ++n)
#pragma unroll
    for (int r = 0; r < 8; ++r)
      out[((size_t)b * Tn + q0 + r + 8 * h) * Dn + n * 16 + lr] = o[n][r] * inv[r];
}

// ---------------------------------------------------------------------------
extern "C" void kernel_launch(void* const* d_in, const int* in_sizes, int n_in,
                              void* d_out, int out_size, void* d_ws, size_t ws_size,
                              hipStream_t stream) {
  (void)in_sizes; (void)n_in; (void)out_size; (void)ws_size;
  const float* x  = (const float*)d_in[0];
  const float* Wq = (const float*)d_in[1];
  const float* Wk = (const float*)d_in[2];
  const float* Wv = (const float*)d_in[3];
  float* out = (float*)d_out;

  char* ws = (char*)d_ws;
  size_t off = 0;
  unsigned short* xb  = (unsigned short*)(ws + off); off += (size_t)Bn * Tn * En * 2;
  unsigned short* wqb = (unsigned short*)(ws + off); off += (size_t)Dn * En * 2;
  unsigned short* wkb = (unsigned short*)(ws + off); off += (size_t)Dn * En * 2;
  unsigned short* wvb = (unsigned short*)(ws + off); off += (size_t)Dn * En * 2;
  unsigned short* qbf = (unsigned short*)(ws + off); off += (size_t)Bn * Tn * Dn * 2;
  unsigned short* kbf = (unsigned short*)(ws + off); off += (size_t)Bn * Tn * Dn * 2;
  unsigned short* vtb = (unsigned short*)(ws + off); off += (size_t)Bn * Tn * Dn * 2;

  const int nx4 = Bn * Tn * En / 4;
  cvt_f32_to_bf16<<<(nx4 + 255) / 256, 256, 0, stream>>>(x, xb, nx4);
  const int nw4 = Dn * En / 4;
  cvt_f32_to_bf16<<<(nw4 + 255) / 256, 256, 0, stream>>>(Wq, wqb, nw4);
  cvt_f32_to_bf16<<<(nw4 + 255) / 256, 256, 0, stream>>>(Wk, wkb, nw4);
  cvt_f32_to_bf16<<<(nw4 + 255) / 256, 256, 0, stream>>>(Wv, wvb, nw4);

  dim3 g1(Bn * Tn / 16 / 8, 3);
  qkv_gemm_kernel<<<g1, 256, 0, stream>>>(xb, wqb, wkb, wvb, qbf, kbf, vtb);

  flash_attn_kernel<<<Bn * Tn / 16, 32, 0, stream>>>(qbf, kbf, vtb, out);
}
